// VAT_CNN_attention_1D_89343909691498
// MI455X (gfx1250) — compile-verified, matrix operand order checked by reading, not run on hardware
//
#include <hip/hip_runtime.h>
#include <hip/hip_bf16.h>
#include <math.h>

// ---------------------------------------------------------------------------
// Problem dimensions (from reference)
// ---------------------------------------------------------------------------
#define BB    8
#define TT    2560
#define FF    229
#define C1    5          // OF/16
#define C2    11         // OF/8
#define F1    114        // FF/2
#define F2    57         // F1/2
#define MM    (BB*TT)    // 20480 rows
#define FEAT  (C2*F2)    // 627
#define KP1   640        // FEAT padded to mult of 32
#define NP1   96         // OF padded to mult of 16
#define OFD   88
#define MCD   48
#define QKVN  144        // 3*MC, already mult of 16
#define KP2   96         // OF padded to 32
#define GG    8
#define KW    31
#define DH    6
#define PADW  15
#define BN_EPS 1e-5f

#define CDIV(a,b) (((a)+(b)-1)/(b))

typedef __attribute__((ext_vector_type(16))) _Float16 v16h;
typedef __attribute__((ext_vector_type(8)))  _Float16 v8h;
typedef __attribute__((ext_vector_type(8)))  float    v8f;

// ---------------------------------------------------------------------------
// Conv1 (1 -> C1) + BN + ReLU
// ---------------------------------------------------------------------------
__global__ void k_conv1(const float* __restrict__ spec, const float* __restrict__ w,
                        const float* __restrict__ bias, const float* __restrict__ g,
                        const float* __restrict__ bb, const float* __restrict__ m,
                        const float* __restrict__ v, float* __restrict__ out) {
    long n = (long)blockIdx.x * blockDim.x + threadIdx.x;
    const long total = (long)BB * C1 * TT * FF;
    if (n >= total) return;
    int f = (int)(n % FF);
    int t = (int)((n / FF) % TT);
    int c = (int)((n / ((long)FF * TT)) % C1);
    int b = (int)(n / ((long)FF * TT * C1));

    float acc = bias[c];
    #pragma unroll
    for (int dt = 0; dt < 3; ++dt) {
        int tt = t + dt - 1;
        if (tt < 0 || tt >= TT) continue;
        #pragma unroll
        for (int df = 0; df < 3; ++df) {
            int ff = f + df - 1;
            if (ff < 0 || ff >= FF) continue;
            acc += w[c * 9 + dt * 3 + df] * spec[((long)b * TT + tt) * FF + ff];
        }
    }
    float scale = g[c] * rsqrtf(v[c] + BN_EPS);
    float shift = bb[c] - m[c] * scale;
    out[(((long)b * C1 + c) * TT + t) * FF + f] = fmaxf(acc * scale + shift, 0.f);
}

// ---------------------------------------------------------------------------
// Conv2 (C1 -> C1) + BN + ReLU + maxpool(width 2)
// ---------------------------------------------------------------------------
__global__ void k_conv2_pool(const float* __restrict__ in, const float* __restrict__ w,
                             const float* __restrict__ bias, const float* __restrict__ g,
                             const float* __restrict__ bb, const float* __restrict__ m,
                             const float* __restrict__ v, float* __restrict__ out) {
    long n = (long)blockIdx.x * blockDim.x + threadIdx.x;
    const long total = (long)BB * C1 * TT * F1;
    if (n >= total) return;
    int fp = (int)(n % F1);
    int t  = (int)((n / F1) % TT);
    int c  = (int)((n / ((long)F1 * TT)) % C1);
    int b  = (int)(n / ((long)F1 * TT * C1));

    float scale = g[c] * rsqrtf(v[c] + BN_EPS);
    float shift = bb[c] - m[c] * scale;
    float best = 0.f;
    #pragma unroll
    for (int s = 0; s < 2; ++s) {
        int f = 2 * fp + s;
        float acc = bias[c];
        for (int ci = 0; ci < C1; ++ci) {
            #pragma unroll
            for (int dt = 0; dt < 3; ++dt) {
                int tt = t + dt - 1;
                if (tt < 0 || tt >= TT) continue;
                #pragma unroll
                for (int df = 0; df < 3; ++df) {
                    int ff = f + df - 1;
                    if (ff < 0 || ff >= FF) continue;
                    acc += w[((c * C1 + ci) * 3 + dt) * 3 + df] *
                           in[(((long)b * C1 + ci) * TT + tt) * FF + ff];
                }
            }
        }
        float r = fmaxf(acc * scale + shift, 0.f);
        best = fmaxf(best, r);
    }
    out[(((long)b * C1 + c) * TT + t) * F1 + fp] = best;
}

// ---------------------------------------------------------------------------
// Conv3 (C1 -> C2) + BN + ReLU + maxpool(width 2)
// ---------------------------------------------------------------------------
__global__ void k_conv3_pool(const float* __restrict__ in, const float* __restrict__ w,
                             const float* __restrict__ bias, const float* __restrict__ g,
                             const float* __restrict__ bb, const float* __restrict__ m,
                             const float* __restrict__ v, float* __restrict__ out) {
    long n = (long)blockIdx.x * blockDim.x + threadIdx.x;
    const long total = (long)BB * C2 * TT * F2;
    if (n >= total) return;
    int fp = (int)(n % F2);
    int t  = (int)((n / F2) % TT);
    int c  = (int)((n / ((long)F2 * TT)) % C2);
    int b  = (int)(n / ((long)F2 * TT * C2));

    float scale = g[c] * rsqrtf(v[c] + BN_EPS);
    float shift = bb[c] - m[c] * scale;
    float best = 0.f;
    #pragma unroll
    for (int s = 0; s < 2; ++s) {
        int f = 2 * fp + s;
        float acc = bias[c];
        for (int ci = 0; ci < C1; ++ci) {
            #pragma unroll
            for (int dt = 0; dt < 3; ++dt) {
                int tt = t + dt - 1;
                if (tt < 0 || tt >= TT) continue;
                #pragma unroll
                for (int df = 0; df < 3; ++df) {
                    int ff = f + df - 1;
                    if (ff < 0 || ff >= F1) continue;
                    acc += w[((c * C1 + ci) * 3 + dt) * 3 + df] *
                           in[(((long)b * C1 + ci) * TT + tt) * F1 + ff];
                }
            }
        }
        float r = fmaxf(acc * scale + shift, 0.f);
        best = fmaxf(best, r);
    }
    out[(((long)b * C2 + c) * TT + t) * F2 + fp] = best;
}

// ---------------------------------------------------------------------------
// Gather conv3 output into (M, KP1) f16 feature matrix, zero-padded.
// feature index kk = c*F2 + fp  (matches transpose(0,2,1,3).reshape)
// ---------------------------------------------------------------------------
__global__ void k_feat_cast(const float* __restrict__ x3, _Float16* __restrict__ feat) {
    long n = (long)blockIdx.x * blockDim.x + threadIdx.x;
    const long total = (long)MM * KP1;
    if (n >= total) return;
    int kk = (int)(n % KP1);
    long row = n / KP1;
    int t = (int)(row % TT);
    int b = (int)(row / TT);
    float val = 0.f;
    if (kk < FEAT) {
        int c = kk / F2, fp = kk % F2;
        val = x3[(((long)b * C2 + c) * TT + t) * F2 + fp];
    }
    feat[n] = (_Float16)val;
}

// ---------------------------------------------------------------------------
// Generic f32 -> f16 cast with zero padding: src (Rreal,Creal) -> dst (Rpad,Cpad)
// ---------------------------------------------------------------------------
__global__ void k_pad_cast(_Float16* __restrict__ dst, const float* __restrict__ src,
                           int Rreal, int Creal, int Rpad, int Cpad) {
    long n = (long)blockIdx.x * blockDim.x + threadIdx.x;
    const long total = (long)Rpad * Cpad;
    if (n >= total) return;
    int c = (int)(n % Cpad);
    int r = (int)(n / Cpad);
    float val = (r < Rreal && c < Creal) ? src[(long)r * Creal + c] : 0.f;
    dst[n] = (_Float16)val;
}

// ---------------------------------------------------------------------------
// WMMA GEMM: C(M x Nreal) = A(M x Kpad) * B(Npad x Kpad)^T  (+ bias)
// A, B in f16, row-major with leading dim = Kpad. One wave per 16x16 tile.
// Fragment layout per CDNA5 ISA 16-bit A-matrix: lanes 0-15 take k-offsets
// {0..7,16..23}, lanes 16-31 take {8..15,24..31} => two b128 loads per frag.
// ---------------------------------------------------------------------------
__global__ void k_wmma_gemm(const _Float16* __restrict__ A, const _Float16* __restrict__ Bm,
                            float* __restrict__ C, const float* __restrict__ bias,
                            int Mrows, int Nreal, int Kpad, int ldc, int tilesN) {
    int wid  = threadIdx.x >> 5;
    int lane = threadIdx.x & 31;
    int tile = blockIdx.x * (blockDim.x >> 5) + wid;
    int tilesM = Mrows >> 4;
    if (tile >= tilesM * tilesN) return;   // uniform per wave: EXEC stays all-1

    int mt = tile / tilesN;
    int nt = tile % tilesN;
    int mrow = (mt << 4) + (lane & 15);
    int nrow = (nt << 4) + (lane & 15);
    int koff = (lane < 16) ? 0 : 8;

    const _Float16* pa = A  + (long)mrow * Kpad + koff;
    const _Float16* pb = Bm + (long)nrow * Kpad + koff;

    v8f acc = {};
    for (int k0 = 0; k0 < Kpad; k0 += 32) {
        union { v16h v; v8h h[2]; } af, bf;
        af.h[0] = *(const v8h*)(pa + k0);
        af.h[1] = *(const v8h*)(pa + k0 + 16);
        bf.h[0] = *(const v8h*)(pb + k0);
        bf.h[1] = *(const v8h*)(pb + k0 + 16);
        __builtin_prefetch(pa + k0 + 32, 0, 1);   // global_prefetch_b8
        acc = __builtin_amdgcn_wmma_f32_16x16x32_f16(
                  /*neg_a=*/false, af.v, /*neg_b=*/false, bf.v,
                  /*c_mod=*/(short)0, acc, /*reuse_a=*/false, /*reuse_b=*/false);
    }

    // C/D layout: VGPR i, lanes 0-15 -> M = i, lanes 16-31 -> M = 8+i; N = lane%16
    int row0 = (mt << 4) + ((lane >> 4) << 3);
    int col  = (nt << 4) + (lane & 15);
    if (col < Nreal) {
        float bv = bias ? bias[col] : 0.f;
        #pragma unroll
        for (int i = 0; i < 8; ++i)
            C[(long)(row0 + i) * ldc + col] = acc[i] + bv;
    }
}

// ---------------------------------------------------------------------------
// Local-window grouped attention. One thread per (b,t,g).
// qkv: (M,144) f32  [0:48)=q, [48:96)=k, [96:144)=v.  rel: (MC,KW).
// Out-of-range window rows: k contribution is 0 but q.rel still applies.
// ---------------------------------------------------------------------------
__global__ void k_attn(const float* __restrict__ qkv, const float* __restrict__ rel,
                       float* __restrict__ ctx, float* __restrict__ a_out) {
    long n = (long)blockIdx.x * blockDim.x + threadIdx.x;
    const long total = (long)BB * TT * GG;
    if (n >= total) return;
    int g = (int)(n % GG);
    int t = (int)((n / GG) % TT);
    int b = (int)(n / ((long)GG * TT));
    long row = (long)b * TT + t;

    float q[DH];
    #pragma unroll
    for (int d = 0; d < DH; ++d) q[d] = qkv[row * QKVN + g * DH + d];

    float e[KW];
    float emax = -3.0e38f;
    for (int j = 0; j < KW; ++j) {
        int tt = t - PADW + j;
        float s = 0.f;
        if (tt >= 0 && tt < TT) {
            long r2 = (long)b * TT + tt;
            #pragma unroll
            for (int d = 0; d < DH; ++d)
                s += q[d] * (qkv[r2 * QKVN + MCD + g * DH + d] + rel[(g * DH + d) * KW + j]);
        } else {
            #pragma unroll
            for (int d = 0; d < DH; ++d)
                s += q[d] * rel[(g * DH + d) * KW + j];
        }
        e[j] = s;
        emax = fmaxf(emax, s);
    }
    float esum = 0.f;
    for (int j = 0; j < KW; ++j) { e[j] = __expf(e[j] - emax); esum += e[j]; }
    float inv = 1.f / esum;

    float o[DH] = {0.f, 0.f, 0.f, 0.f, 0.f, 0.f};
    for (int j = 0; j < KW; ++j) {
        float wgt = e[j] * inv;
        a_out[(row * GG + g) * KW + j] = wgt;
        int tt = t - PADW + j;
        if (tt >= 0 && tt < TT) {
            long r2 = (long)b * TT + tt;
            #pragma unroll
            for (int d = 0; d < DH; ++d)
                o[d] += wgt * qkv[r2 * QKVN + 2 * MCD + g * DH + d];
        }
    }
    #pragma unroll
    for (int d = 0; d < DH; ++d) ctx[row * MCD + g * DH + d] = o[d];
}

// ---------------------------------------------------------------------------
// LayerNorm over MC=48. One thread per row.
// ---------------------------------------------------------------------------
__global__ void k_layernorm(const float* __restrict__ ctx, const float* __restrict__ lg,
                            const float* __restrict__ lb, float* __restrict__ h) {
    long row = (long)blockIdx.x * blockDim.x + threadIdx.x;
    if (row >= MM) return;
    float v[MCD];
    float mu = 0.f;
    #pragma unroll
    for (int c = 0; c < MCD; ++c) { v[c] = ctx[row * MCD + c]; mu += v[c]; }
    mu *= (1.f / MCD);
    float var = 0.f;
    #pragma unroll
    for (int c = 0; c < MCD; ++c) { float d = v[c] - mu; var += d * d; }
    var *= (1.f / MCD);
    float inv = rsqrtf(var + 1e-5f);
    #pragma unroll
    for (int c = 0; c < MCD; ++c)
        h[row * MCD + c] = (v[c] - mu) * inv * lg[c] + lb[c];
}

// ---------------------------------------------------------------------------
// Final linear (48 -> 88) + sigmoid. One thread per output element.
// ---------------------------------------------------------------------------
__global__ void k_lin_sigmoid(const float* __restrict__ h, const float* __restrict__ w,
                              const float* __restrict__ bias, float* __restrict__ out) {
    long n = (long)blockIdx.x * blockDim.x + threadIdx.x;
    const long total = (long)MM * OFD;
    if (n >= total) return;
    int o = (int)(n % OFD);
    long row = n / OFD;
    float s = bias[o];
    #pragma unroll
    for (int c = 0; c < MCD; ++c)
        s += h[row * MCD + c] * w[o * MCD + c];
    out[n] = 1.f / (1.f + __expf(-s));
}

// ---------------------------------------------------------------------------
// Launcher
// ---------------------------------------------------------------------------
extern "C" void kernel_launch(void* const* d_in, const int* in_sizes, int n_in,
                              void* d_out, int out_size, void* d_ws, size_t ws_size,
                              hipStream_t stream) {
    const float* spec  = (const float*)d_in[0];
    const float* c1w   = (const float*)d_in[1];
    const float* c1b   = (const float*)d_in[2];
    const float* bn1g  = (const float*)d_in[3];
    const float* bn1b  = (const float*)d_in[4];
    const float* bn1m  = (const float*)d_in[5];
    const float* bn1v  = (const float*)d_in[6];
    const float* c2w   = (const float*)d_in[7];
    const float* c2b   = (const float*)d_in[8];
    const float* bn2g  = (const float*)d_in[9];
    const float* bn2b  = (const float*)d_in[10];
    const float* bn2m  = (const float*)d_in[11];
    const float* bn2v  = (const float*)d_in[12];
    const float* c3w   = (const float*)d_in[13];
    const float* c3b   = (const float*)d_in[14];
    const float* bn3g  = (const float*)d_in[15];
    const float* bn3b  = (const float*)d_in[16];
    const float* bn3m  = (const float*)d_in[17];
    const float* bn3v  = (const float*)d_in[18];
    const float* fcw   = (const float*)d_in[19];
    const float* fcb   = (const float*)d_in[20];
    const float* wq    = (const float*)d_in[21];
    const float* wk    = (const float*)d_in[22];
    const float* wv    = (const float*)d_in[23];
    const float* rel   = (const float*)d_in[24];
    const float* lng   = (const float*)d_in[25];
    const float* lnb   = (const float*)d_in[26];
    const float* linw  = (const float*)d_in[27];
    const float* linb  = (const float*)d_in[28];

    // ---- workspace layout (bump allocation with region reuse) --------------
    char* ws = (char*)d_ws;
    const size_t szX1   = (size_t)BB * C1 * TT * FF * 4;      // 93,798,400 (region A)
    const size_t szX2   = (size_t)BB * C1 * TT * F1 * 4;      // 46,694,400 (region B)
    const size_t offA   = 0;
    const size_t offB   = szX1;                                // 256-aligned
    const size_t offC   = offB + szX2;

    float*    x1      = (float*)(ws + offA);                   // conv1 out
    float*    x2      = (float*)(ws + offB);                   // conv2+pool out
    float*    x3      = (float*)(ws + offA);                   // reuses region A
    _Float16* feat_h  = (_Float16*)(ws + offB);                // reuses region B
    _Float16* fcw_h   = (_Float16*)(ws + offB + (size_t)MM * KP1 * 2);
    _Float16* wqkv_h  = (_Float16*)((char*)fcw_h + (size_t)NP1 * KP1 * 2);

    char* pc = ws + offC;
    float*    xfc     = (float*)pc;        pc += (size_t)MM * OFD * 4;
    _Float16* xh      = (_Float16*)pc;     pc += (size_t)MM * KP2 * 2;
    float*    qkv     = (float*)pc;        pc += (size_t)MM * QKVN * 4;
    float*    ctx     = (float*)pc;        pc += (size_t)MM * MCD * 4;
    float*    hnorm   = (float*)pc;        pc += (size_t)MM * MCD * 4;

    float* frame_pred = (float*)d_out;
    float* a_out      = frame_pred + (size_t)MM * OFD;

    const int TPB = 256;

    // 1) conv stack
    k_conv1<<<CDIV((long)BB*C1*TT*FF, TPB), TPB, 0, stream>>>(
        spec, c1w, c1b, bn1g, bn1b, bn1m, bn1v, x1);
    k_conv2_pool<<<CDIV((long)BB*C1*TT*F1, TPB), TPB, 0, stream>>>(
        x1, c2w, c2b, bn2g, bn2b, bn2m, bn2v, x2);
    k_conv3_pool<<<CDIV((long)BB*C2*TT*F2, TPB), TPB, 0, stream>>>(
        x2, c3w, c3b, bn3g, bn3b, bn3m, bn3v, x3);

    // 2) cast operands for WMMA GEMMs (pads rewritten every call)
    k_feat_cast<<<CDIV((long)MM*KP1, TPB), TPB, 0, stream>>>(x3, feat_h);
    k_pad_cast<<<CDIV((long)NP1*KP1, TPB), TPB, 0, stream>>>(fcw_h, fcw, OFD, FEAT, NP1, KP1);
    k_pad_cast<<<CDIV((long)MCD*KP2, TPB), TPB, 0, stream>>>(wqkv_h,             wq, MCD, OFD, MCD, KP2);
    k_pad_cast<<<CDIV((long)MCD*KP2, TPB), TPB, 0, stream>>>(wqkv_h +   MCD*KP2, wk, MCD, OFD, MCD, KP2);
    k_pad_cast<<<CDIV((long)MCD*KP2, TPB), TPB, 0, stream>>>(wqkv_h + 2*MCD*KP2, wv, MCD, OFD, MCD, KP2);

    // 3) FC GEMM: (20480 x 640) x (96 x 640)^T -> (20480 x 88), bias fused
    {
        int tilesN = NP1 / 16;                       // 6
        int tiles  = (MM / 16) * tilesN;             // 7680 waves
        k_wmma_gemm<<<CDIV(tiles, TPB/32), TPB, 0, stream>>>(
            feat_h, fcw_h, xfc, fcb, MM, OFD, KP1, OFD, tilesN);
    }

    // 4) cast FC output to f16 (20480 x 96, zero-padded)
    k_pad_cast<<<CDIV((long)MM*KP2, TPB), TPB, 0, stream>>>(xh, xfc, MM, OFD, MM, KP2);

    // 5) QKV GEMM: (20480 x 96) x (144 x 96)^T -> (20480 x 144)
    {
        int tilesN = QKVN / 16;                      // 9
        int tiles  = (MM / 16) * tilesN;             // 11520 waves
        k_wmma_gemm<<<CDIV(tiles, TPB/32), TPB, 0, stream>>>(
            xh, wqkv_h, qkv, nullptr, MM, QKVN, KP2, QKVN, tilesN);
    }

    // 6) windowed grouped attention (also emits `a`)
    k_attn<<<CDIV((long)BB*TT*GG, TPB), TPB, 0, stream>>>(qkv, rel, ctx, a_out);

    // 7) layernorm + final linear + sigmoid
    k_layernorm<<<CDIV(MM, TPB), TPB, 0, stream>>>(ctx, lng, lnb, hnorm);
    k_lin_sigmoid<<<CDIV((long)MM*OFD, TPB), TPB, 0, stream>>>(hnorm, linw, linb, frame_pred);
}
